// selfAttention_11647951306946
// MI455X (gfx1250) — matrix operand order repaired and checked by reading, not compiled, not run on hardware
//
#include <hip/hip_runtime.h>
#include <hip/hip_bf16.h>

// ---------------------------------------------------------------------------
// Self-attention for MI455X (gfx1250), wave32, WMMA f32_16x16x32_f16.
//   N=4096 tokens, D=1024 dims.
//   k1: QKV projection (f32 in -> f16 Q,K row-major; V transposed f16)
//   k2: S = scale * Q @ K^T      (f16 WMMA, f32 out)   [async-LDS double buffer]
//   k3: row softmax, P written f16 in-place over S (row stride 2N halves)
//   k4: O = P @ Vt^T(feat-major) (f16 WMMA, f32 out)   [async-LDS double buffer]
// ---------------------------------------------------------------------------

#define N_TOK 4096
#define DMODEL 1024

typedef __attribute__((ext_vector_type(16))) _Float16 v16h;
typedef __attribute__((ext_vector_type(8)))  _Float16 v8h;
typedef __attribute__((ext_vector_type(4)))  _Float16 v4h;
typedef __attribute__((ext_vector_type(8)))  float    v8f;
typedef int v4i __attribute__((vector_size(16)));   // async b128 payload type

constexpr int BM = 128;   // block tile M
constexpr int BN = 128;   // block tile N
constexpr int BK = 32;    // K step (== WMMA K)
constexpr int PITCH = 40; // LDS row pitch in halves (80B, conflict-avoiding)

#define AS1 __attribute__((address_space(1)))
#define AS3 __attribute__((address_space(3)))

#ifndef __has_builtin
#define __has_builtin(x) 0
#endif

#if __has_builtin(__builtin_amdgcn_global_load_async_to_lds_b128)
#define USE_ASYNC 1
#else
#define USE_ASYNC 0
#endif

__device__ inline void async_wait0() {
#if __has_builtin(__builtin_amdgcn_s_wait_asynccnt)
  __builtin_amdgcn_s_wait_asynccnt(0);
#else
  asm volatile("s_wait_asynccnt 0" ::: "memory");
#endif
}

__device__ inline v16h cat16(v8h lo, v8h hi) {
  return __builtin_shufflevector(lo, hi, 0,1,2,3,4,5,6,7,8,9,10,11,12,13,14,15);
}

__device__ inline v8f wmma_f16(v16h a, v16h b, v8f c) {
  // D = A(16x32 f16) * B(32x16 f16) + C(16x16 f32)
  return __builtin_amdgcn_wmma_f32_16x16x32_f16(
      /*neg_a=*/false, a, /*neg_b=*/false, b,
      /*c_mod=*/(short)0, c, /*reuse_a=*/false, /*reuse_b=*/false);
}

// A-fragment: 16x32 f16. Lane group g=lane>>4 holds K = {8g..8g+7, 16+8g..16+8g+7}
// for row m = rowBase + (lane&15). Tile in LDS is [row][k] with pitch PITCH.
__device__ inline v16h load_fragA(const _Float16* t, int rowBase, int lane) {
  const int m = rowBase + (lane & 15);
  const int g = lane >> 4;
  const _Float16* p = t + m * PITCH + g * 8;
  v8h lo = *(const v8h*)p;          // K = 8g .. 8g+7
  v8h hi = *(const v8h*)(p + 16);   // K = 16+8g .. 16+8g+7
  return cat16(lo, hi);
}

// B-fragment: 32x16 f16. Lane group g holds K = {8g..8g+7, 16+8g..16+8g+7} for
// output column n = colBase + (lane&15). LDS tile is [n][k] (i.e. B^T row-major).
__device__ inline v16h load_fragB(const _Float16* t, int colBase, int lane) {
  const int n = colBase + (lane & 15);
  const int g = lane >> 4;
  const _Float16* p = t + n * PITCH + g * 8;
  v8h lo = *(const v8h*)p;
  v8h hi = *(const v8h*)(p + 16);
  return cat16(lo, hi);
}

// Stage a 128x32 tile of f32 data into LDS as f16 (rows mb.., cols kb..kb+31).
__device__ inline void stage_f32(const float* __restrict__ src, size_t ld,
                                 int mb, int kb, _Float16* lds, int tid) {
#pragma unroll
  for (int p = 0; p < 4; ++p) {
    const int idx = p * 256 + tid;       // float4 units, 0..1023
    const int row = idx >> 3;            // 8 float4 per 32-float row
    const int c   = (idx & 7) * 4;
    const float4 f = *(const float4*)(src + (size_t)(mb + row) * ld + kb + c);
    v4h h = { (_Float16)f.x, (_Float16)f.y, (_Float16)f.z, (_Float16)f.w };
    *(v4h*)(lds + row * PITCH + c) = h;
  }
}

// Stage a 128x32 tile of f16 data into LDS (rows mb.., cols kb..kb+31).
__device__ inline void stage_f16(const _Float16* __restrict__ src, size_t ld,
                                 int mb, int kb, _Float16* lds, int tid) {
#pragma unroll
  for (int p = 0; p < 2; ++p) {
    const int idx = p * 256 + tid;       // v8h units, 0..511
    const int row = idx >> 2;            // 4 units per 32-half row
    const int u   = (idx & 3) * 8;
    v8h v = *(const v8h*)(src + (size_t)(mb + row) * ld + kb + u);
    *(v8h*)(lds + row * PITCH + u) = v;
  }
}

#if USE_ASYNC
// Async DMA global f16 tile -> LDS (no VGPR round trip, tracked by ASYNCcnt).
__device__ inline void stage_f16_async(const _Float16* __restrict__ src, size_t ld,
                                       int mb, int kb, _Float16* lds, int tid) {
#pragma unroll
  for (int p = 0; p < 2; ++p) {
    const int idx = p * 256 + tid;       // 16B units, 0..511
    const int row = idx >> 2;
    const int u   = (idx & 3) * 8;
    AS1 v4i* g = (AS1 v4i*)(src + (size_t)(mb + row) * ld + kb + u);
    AS3 v4i* l = (AS3 v4i*)(lds + row * PITCH + u);
    __builtin_amdgcn_global_load_async_to_lds_b128(g, l, 0, 0);
  }
}
#endif

// One BK step: wave computes its 32x64 sub-tile (2 M-tiles x 4 N-tiles).
__device__ inline void mma_step(const _Float16* lA, const _Float16* lB,
                                int wm0, int wn0, int lane, v8f acc[2][4]) {
  v16h a0 = load_fragA(lA, wm0,      lane);
  v16h a1 = load_fragA(lA, wm0 + 16, lane);
  v16h b[4];
#pragma unroll
  for (int j = 0; j < 4; ++j) b[j] = load_fragB(lB, wn0 + 16 * j, lane);
#pragma unroll
  for (int j = 0; j < 4; ++j) {
    acc[0][j] = wmma_f16(a0, b[j], acc[0][j]);
    acc[1][j] = wmma_f16(a1, b[j], acc[1][j]);
  }
}

// ------------------------- Kernel 1: QKV projection -------------------------
// C[m][n] = sum_d X[m][d] * W[n][d]   (W stored [d_out][d_in] -> B^T rows!)
// z=0 -> Q f16 [tok][feat], z=1 -> K f16 [tok][feat], z=2 -> Vt f16 [feat][tok]
__global__ __launch_bounds__(256) void qkv_kernel(
    const float* __restrict__ x,
    const float* __restrict__ wq, const float* __restrict__ wk,
    const float* __restrict__ wv,
    _Float16* __restrict__ Qh, _Float16* __restrict__ Kh,
    _Float16* __restrict__ Vt) {
  __shared__ _Float16 lA[BM * PITCH];
  __shared__ _Float16 lB[BN * PITCH];
  const int tid = threadIdx.x, lane = tid & 31, wave = tid >> 5;
  const int wm0 = (wave & 3) * 32, wn0 = (wave >> 2) * 64;
  const int mb = blockIdx.y * BM, nb = blockIdx.x * BN;
  const int z = blockIdx.z;
  const float* w = (z == 0) ? wq : (z == 1) ? wk : wv;

  v8f acc[2][4];
#pragma unroll
  for (int i = 0; i < 2; ++i)
#pragma unroll
    for (int j = 0; j < 4; ++j)
      acc[i][j] = (v8f){0.f,0.f,0.f,0.f,0.f,0.f,0.f,0.f};

  for (int kb = 0; kb < DMODEL; kb += BK) {
    __syncthreads();
    stage_f32(x, DMODEL, mb, kb, lA, tid);
    stage_f32(w, DMODEL, nb, kb, lB, tid);
    if (kb + BK < DMODEL) {
      __builtin_prefetch(x + (size_t)(mb + (tid >> 1)) * DMODEL + kb + BK);
      __builtin_prefetch(w + (size_t)(nb + (tid >> 1)) * DMODEL + kb + BK);
    }
    __syncthreads();
    mma_step(lA, lB, wm0, wn0, lane, acc);
  }

  const int g = lane >> 4, n15 = lane & 15;
  if (z < 2) {
    _Float16* out = (z == 0) ? Qh : Kh;   // [tok][feat], ld = DMODEL
#pragma unroll
    for (int i = 0; i < 2; ++i)
#pragma unroll
      for (int j = 0; j < 4; ++j) {
        const int feat = nb + wn0 + 16 * j + n15;
#pragma unroll
        for (int r = 0; r < 8; ++r) {
          const int tok = mb + wm0 + 16 * i + 8 * g + r;
          out[(size_t)tok * DMODEL + feat] = (_Float16)acc[i][j][r];
        }
      }
  } else {
    // Vt[feat][tok]: lane owns one feature column, 8 contiguous tokens -> b128
#pragma unroll
    for (int i = 0; i < 2; ++i)
#pragma unroll
      for (int j = 0; j < 4; ++j) {
        const int feat = nb + wn0 + 16 * j + n15;
        const int tok0 = mb + wm0 + 16 * i + 8 * g;
        v8h h;
#pragma unroll
        for (int r = 0; r < 8; ++r) h[r] = (_Float16)acc[i][j][r];
        *(v8h*)(Vt + (size_t)feat * N_TOK + tok0) = h;
      }
  }
}

// --------------------- Kernel 2/4: generic f16 GEMM -------------------------
// C[m][n] = scale * sum_k A[m][k] * B[n][k], A/B f16 row-major-over-K, C f32.
// Double-buffered LDS; async global->LDS DMA when available.
__global__ __launch_bounds__(256) void gemm_f16_kernel(
    const _Float16* __restrict__ A, long ldA,
    const _Float16* __restrict__ B, long ldB,
    int kdim, float* __restrict__ out, long ldO, float scale) {
  __shared__ _Float16 lA[2][BM * PITCH];
  __shared__ _Float16 lB[2][BN * PITCH];
  const int tid = threadIdx.x, lane = tid & 31, wave = tid >> 5;
  const int wm0 = (wave & 3) * 32, wn0 = (wave >> 2) * 64;
  const int mb = blockIdx.y * BM, nb = blockIdx.x * BN;

  v8f acc[2][4];
#pragma unroll
  for (int i = 0; i < 2; ++i)
#pragma unroll
    for (int j = 0; j < 4; ++j)
      acc[i][j] = (v8f){0.f,0.f,0.f,0.f,0.f,0.f,0.f,0.f};

#if USE_ASYNC
  // Prologue: DMA first tile pair into buffer 0.
  stage_f16_async(A, (size_t)ldA, mb, 0, lA[0], tid);
  stage_f16_async(B, (size_t)ldB, nb, 0, lB[0], tid);
  async_wait0();
  __syncthreads();
  int cur = 0;
  for (int kb = 0; kb < kdim; kb += BK) {
    const int nxt = cur ^ 1;
    if (kb + BK < kdim) {
      // DMA next tile pair while WMMAs below consume the current one.
      stage_f16_async(A, (size_t)ldA, mb, kb + BK, lA[nxt], tid);
      stage_f16_async(B, (size_t)ldB, nb, kb + BK, lB[nxt], tid);
    }
    mma_step(lA[cur], lB[cur], wm0, wn0, lane, acc);
    async_wait0();     // next tile landed (own wave's DMA)
    __syncthreads();   // all waves done reading cur / writing nxt
    cur = nxt;
  }
#else
  for (int kb = 0; kb < kdim; kb += BK) {
    __syncthreads();
    stage_f16(A, (size_t)ldA, mb, kb, lA[0], tid);
    stage_f16(B, (size_t)ldB, nb, kb, lB[0], tid);
    if (kb + BK < kdim) {
      __builtin_prefetch(A + (size_t)(mb + (tid >> 1)) * ldA + kb + BK);
      __builtin_prefetch(B + (size_t)(nb + (tid >> 1)) * ldB + kb + BK);
    }
    __syncthreads();
    mma_step(lA[0], lB[0], wm0, wn0, lane, acc);
  }
#endif

  const int g = lane >> 4, n15 = lane & 15;
#pragma unroll
  for (int i = 0; i < 2; ++i)
#pragma unroll
    for (int j = 0; j < 4; ++j) {
      const int n = nb + wn0 + 16 * j + n15;
#pragma unroll
      for (int r = 0; r < 8; ++r) {
        const int m = mb + wm0 + 16 * i + 8 * g + r;
        out[(size_t)m * ldO + n] = acc[i][j][r] * scale;
      }
    }
}

// ------------------------- Kernel 3: row softmax ----------------------------
// One block per row. Reads f32 row of S, writes normalized probabilities as
// f16 IN-PLACE over the same bytes (row stride in halves = 2*N_TOK).
__global__ __launch_bounds__(256) void softmax_kernel(float* __restrict__ S) {
  __shared__ float red[256];
  const int row = blockIdx.x, tid = threadIdx.x;
  float* srow = S + (size_t)row * N_TOK;
  _Float16* prow = (_Float16*)srow;  // same bytes, first N_TOK halves used

  float v[16];
  float mx = -3.4e38f;
#pragma unroll
  for (int e = 0; e < 16; ++e) {
    v[e] = srow[tid + 256 * e];
    mx = fmaxf(mx, v[e]);
  }
  red[tid] = mx; __syncthreads();
  for (int s = 128; s > 0; s >>= 1) {
    if (tid < s) red[tid] = fmaxf(red[tid], red[tid + s]);
    __syncthreads();
  }
  mx = red[0]; __syncthreads();

  float sum = 0.f;
#pragma unroll
  for (int e = 0; e < 16; ++e) {
    v[e] = __expf(v[e] - mx);
    sum += v[e];
  }
  red[tid] = sum; __syncthreads();
  for (int s = 128; s > 0; s >>= 1) {
    if (tid < s) red[tid] += red[tid + s];
    __syncthreads();
  }
  const float inv = 1.0f / red[0];
  __syncthreads();  // all reads of this row complete before in-place f16 writes

#pragma unroll
  for (int e = 0; e < 16; ++e)
    prow[tid + 256 * e] = (_Float16)(v[e] * inv);
}

// ---------------------------------------------------------------------------
extern "C" void kernel_launch(void* const* d_in, const int* in_sizes, int n_in,
                              void* d_out, int out_size, void* d_ws, size_t ws_size,
                              hipStream_t stream) {
  const float* x  = (const float*)d_in[0];
  const float* wq = (const float*)d_in[1];
  const float* wk = (const float*)d_in[2];
  const float* wv = (const float*)d_in[3];

  // Workspace: Qh(8MB) | Kh(8MB) | Vt(8MB) | S(64MB f32, reused as f16 P)
  _Float16* Qh = (_Float16*)d_ws;
  _Float16* Kh = Qh + (size_t)N_TOK * DMODEL;
  _Float16* Vt = Kh + (size_t)N_TOK * DMODEL;
  float*    S  = (float*)((char*)d_ws + (size_t)3 * N_TOK * DMODEL * 2);

  // 1) Q, K (f16 row-major), Vt (f16 feature-major)
  qkv_kernel<<<dim3(DMODEL / BN, N_TOK / BM, 3), 256, 0, stream>>>(
      x, wq, wk, wv, Qh, Kh, Vt);

  // 2) S = (1/sqrt(D)) * Q @ K^T   (f32 out)
  gemm_f16_kernel<<<dim3(N_TOK / BN, N_TOK / BM), 256, 0, stream>>>(
      Qh, DMODEL, Kh, DMODEL, DMODEL, S, N_TOK, 0.03125f);

  // 3) row softmax; P f16 written in-place (row stride 2*N_TOK halves)
  softmax_kernel<<<N_TOK, 256, 0, stream>>>(S);

  // 4) O = P @ V  via  C[q][feat] = sum_k P[q][k] * Vt[feat][k]
  gemm_f16_kernel<<<dim3(DMODEL / BN, N_TOK / BM), 256, 0, stream>>>(
      (const _Float16*)S, 2 * N_TOK, Vt, N_TOK, N_TOK, (float*)d_out, DMODEL, 1.0f);
}
